// se3_denoiser_49976239456544
// MI455X (gfx1250) — matrix operand order repaired and checked by reading, not compiled, not run on hardware
//
#include <hip/hip_runtime.h>
#include <hip/hip_bf16.h>
#include <math.h>

// ---------------- problem constants ----------------
#define BGR   16
#define NPG   2048
#define NTOT  32768           // BGR*NPG
#define NEDGE 1048576
#define INC   256
#define DIM   128
#define OUTC  6
#define ENCH  100
#define NATM  20

typedef __attribute__((ext_vector_type(16))) _Float16 v16h;
typedef __attribute__((ext_vector_type(8)))  _Float16 v8h;
typedef __attribute__((ext_vector_type(8)))  float    v8f;

__device__ __forceinline__ float gelu_f(float x) {
    return 0.5f * x * (1.0f + erff(x * 0.70710678118654752f));
}

// A fragment (16x32 f16, row-major source, leading dim ld in elements).
// base points at element (row0, k0).
__device__ __forceinline__ v16h load_a16(const _Float16* base, int ld, int lane) {
    const int row  = lane & 15;
    const int half = lane >> 4;
    const _Float16* p = base + (size_t)row * ld + half * 16;
    v8h lo = *(const v8h*)(p);
    v8h hi = *(const v8h*)(p + 8);
    v16h a;
#pragma unroll
    for (int i = 0; i < 8; ++i) { a[i] = lo[i]; a[i + 8] = hi[i]; }
    return a;
}

// B fragment (32x16 f16) from pre-transposed weights Bt laid out [N][K].
// bt points at element (n0, k0); ldk = K stride of Bt rows.
__device__ __forceinline__ v16h load_b16(const _Float16* bt, int ldk, int lane) {
    const int col  = lane & 15;
    const int half = lane >> 4;
    return *(const v16h*)(bt + (size_t)col * ldk + half * 16);
}

__device__ __forceinline__ v8f wmma_f16(v16h a, v16h b, v8f c) {
    return __builtin_amdgcn_wmma_f32_16x16x32_f16(false, a, false, b, (short)0, c, false, false);
}

// ---------------- K0: weight transpose to f16 [N][K] + zero stats ----------------
__global__ __launch_bounds__(256) void prep_kernel(
    const float* W1, const float* W2, const float* Wm,
    const float* A1, const float* A2, const float* W3,
    _Float16* W1t, _Float16* W2t, _Float16* WmT,
    _Float16* A1t, _Float16* A2t, _Float16* W3t,
    float* gsum, float* gsq) {
    int i = blockIdx.x * blockDim.x + threadIdx.x;
    if (i < 256) { gsum[i] = 0.f; return; }
    if (i < 512) { gsq[i - 256] = 0.f; return; }
    i -= 512;
    if (i < 256 * 256) { int n = i >> 8, k = i & 255; W1t[i] = (_Float16)W1[k * 256 + n]; return; }
    i -= 256 * 256;
    if (i < 128 * 256) { int n = i >> 8, k = i & 255; W2t[i] = (_Float16)W2[k * 128 + n]; return; }
    i -= 128 * 256;
    if (i < 128 * 128) { int n = i >> 7, k = i & 127; WmT[i] = (_Float16)Wm[k * 128 + n]; return; }
    i -= 128 * 128;
    if (i < 128 * 128) { int n = i >> 7, k = i & 127;
        A1t[i] = (n < ENCH) ? (_Float16)A1[k * ENCH + n] : (_Float16)0.f; return; }
    i -= 128 * 128;
    if (i < 16 * 128) { int n = i >> 7, k = i & 127;
        A2t[i] = (n < OUTC && k < ENCH) ? (_Float16)A2[k * OUTC + n] : (_Float16)0.f; return; }
    i -= 16 * 128;
    if (i < 32 * 128) { int n = i >> 7, k = i & 127;
        W3t[i] = (n < NATM) ? (_Float16)W3[k * NATM + n] : (_Float16)0.f; return; }
}

// ---------------- K1: per-channel sum / sumsq ----------------
__global__ __launch_bounds__(256) void colsum_kernel(const float* x, float* gsum, float* gsq) {
    int c = threadIdx.x;
    int r0 = blockIdx.x * 128;
    float s = 0.f, q = 0.f;
    for (int r = 0; r < 128; ++r) {
        float v = x[(size_t)(r0 + r) * INC + c];
        s += v; q += v * v;
    }
    atomicAdd(&gsum[c], s);
    atomicAdd(&gsq[c], q);
}

// ---------------- K2: BN scale/shift ----------------
__global__ void finalize_kernel(const float* gsum, const float* gsq,
                                const float* gamma, const float* beta,
                                float* scl, float* shf) {
    int c = threadIdx.x;
    float mu  = gsum[c] * (1.0f / (float)NTOT);
    float var = gsq[c] * (1.0f / (float)NTOT) - mu * mu;
    float s = gamma[c] * rsqrtf(var + 1e-5f);
    scl[c] = s;
    shf[c] = beta[c] - mu * s;
}

// ---------------- K3: BN apply + cast to f16 ----------------
__global__ __launch_bounds__(256) void bncast_kernel(const float* x, const float* scl,
                                                     const float* shf, _Float16* XN) {
    size_t i = (size_t)blockIdx.x * blockDim.x + threadIdx.x;
    if (i >= (size_t)NTOT * INC) return;
    int c = (int)(i & 255);
    XN[i] = (_Float16)(x[i] * scl[c] + shf[c]);
}

// ---------------- K4: GEMM1 32768x256x256 + gelu -> H1(f16) ----------------
__global__ __launch_bounds__(256) void gemm1_kernel(const _Float16* XN, const _Float16* W1t,
                                                    const float* b1, _Float16* H1) {
    int wave = (blockIdx.x * blockDim.x + threadIdx.x) >> 5;
    int lane = threadIdx.x & 31;
    int r0 = wave * 16;
    v16h a[8];
#pragma unroll
    for (int kk = 0; kk < 8; ++kk) a[kk] = load_a16(XN + (size_t)r0 * 256 + kk * 32, 256, lane);
    const int col = lane & 15, half = lane >> 4;
    for (int nt = 0; nt < 16; ++nt) {
        v8f c = {};
#pragma unroll
        for (int kk = 0; kk < 8; ++kk)
            c = wmma_f16(a[kk], load_b16(W1t + (size_t)(nt * 16) * 256 + kk * 32, 256, lane), c);
        float bias = b1[nt * 16 + col];
#pragma unroll
        for (int v = 0; v < 8; ++v) {
            float x = gelu_f(c[v] + bias);
            H1[(size_t)(r0 + half * 8 + v) * 256 + nt * 16 + col] = (_Float16)x;
        }
    }
}

// ---------------- K5: GEMM2+gelu -> LDS -> GEMM3 -> argmax ids ----------------
__global__ __launch_bounds__(256) void gemm23_argmax_kernel(
    const _Float16* H1, const _Float16* W2t, const float* b2,
    const _Float16* W3t, const float* b3, int* ids) {
    __shared__ __align__(32) _Float16 h2s[8][16 * 128];
    __shared__ float lg[8][16 * 32];
    int wslot = threadIdx.x >> 5;
    int lane  = threadIdx.x & 31;
    int wave  = blockIdx.x * 8 + wslot;
    int r0 = wave * 16;
    const int col = lane & 15, half = lane >> 4;

    v16h a[8];
#pragma unroll
    for (int kk = 0; kk < 8; ++kk) a[kk] = load_a16(H1 + (size_t)r0 * 256 + kk * 32, 256, lane);
    for (int nt = 0; nt < 8; ++nt) {
        v8f c = {};
#pragma unroll
        for (int kk = 0; kk < 8; ++kk)
            c = wmma_f16(a[kk], load_b16(W2t + (size_t)(nt * 16) * 256 + kk * 32, 256, lane), c);
        float bias = b2[nt * 16 + col];
#pragma unroll
        for (int v = 0; v < 8; ++v)
            h2s[wslot][(half * 8 + v) * 128 + nt * 16 + col] = (_Float16)gelu_f(c[v] + bias);
    }
    __syncthreads();
    v16h a2[4];
#pragma unroll
    for (int kk = 0; kk < 4; ++kk) a2[kk] = load_a16(&h2s[wslot][kk * 32], 128, lane);
    for (int nt = 0; nt < 2; ++nt) {
        v8f c = {};
#pragma unroll
        for (int kk = 0; kk < 4; ++kk)
            c = wmma_f16(a2[kk], load_b16(W3t + (size_t)(nt * 16) * 128 + kk * 32, 128, lane), c);
        int gc = nt * 16 + col;
        float bias = (gc < NATM) ? b3[gc] : -1e30f;
#pragma unroll
        for (int v = 0; v < 8; ++v) lg[wslot][(half * 8 + v) * 32 + gc] = c[v] + bias;
    }
    __syncthreads();
    if (lane < 16) {
        const float* row = &lg[wslot][lane * 32];
        float best = row[0]; int bi = 0;
        for (int j = 1; j < NATM; ++j) { float v = row[j]; if (v > best) { best = v; bi = j; } }
        ids[r0 + lane] = bi;
    }
}

// ---------------- K6: embedding gather -> HF (f32) ----------------
__global__ __launch_bounds__(256) void embed_kernel(const int* ids, const float* embed, float* HF) {
    size_t i = (size_t)blockIdx.x * blockDim.x + threadIdx.x;
    if (i >= (size_t)NTOT * DIM) return;
    int row = (int)(i >> 7), c = (int)(i & 127);
    HF[i] = embed[ids[row] * DIM + c];
}

// ---------------- K7: zero adjacency bitmask ----------------
__global__ __launch_bounds__(256) void zero_mask_kernel(uint4* p, int n4) {
    int i = blockIdx.x * blockDim.x + threadIdx.x;
    if (i < n4) p[i] = make_uint4(0u, 0u, 0u, 0u);
}

// ---------------- K8: scatter edges into bitmask (dedups like .max) ----------------
__global__ __launch_bounds__(256) void scatter_kernel(const int* ei, unsigned* mask) {
    int e = blockIdx.x * blockDim.x + threadIdx.x;
    if (e >= NEDGE) return;
    int s = ei[e], d = ei[NEDGE + e];
    int g = s >> 11;
    if ((d >> 11) != g) return;
    int ls = s & 2047, ld = d & 2047;
    atomicOr(&mask[(((size_t)(g << 11) + ls) << 6) + (ld >> 5)], 1u << (ld & 31));
}

// ---------------- K9: sparse aggregate m + (h+m) -> HIN (f16) ----------------
__global__ __launch_bounds__(256) void aggregate_kernel(const unsigned* mask, const float* HF,
                                                        _Float16* HIN) {
    int wave = (blockIdx.x * blockDim.x + threadIdx.x) >> 5;
    int lane = threadIdx.x & 31;
    int row = wave;                       // 0..NTOT-1
    int b = row >> 11;
    const unsigned* mrow = mask + (size_t)row * 64;
    float a0 = 0.f, a1 = 0.f, a2 = 0.f, a3 = 0.f;
    int deg = 0;
    for (int w = 0; w < 64; ++w) {
        unsigned bits = mrow[w];
        deg += __popc(bits);
        while (bits) {
            int t = __ffs((int)bits) - 1;
            bits &= bits - 1;
            const float* hr = HF + ((size_t)((b << 11) + (w << 5) + t) << 7);
            a0 += hr[lane];
            a1 += hr[lane + 32];
            a2 += hr[lane + 64];
            a3 += hr[lane + 96];
        }
    }
    float dg = (deg > 0) ? (float)deg : 1.0f;
    float inv = 1.0f / dg;
    const float* hs = HF + ((size_t)row << 7);
    _Float16* o = HIN + ((size_t)row << 7);
    o[lane]       = (_Float16)(hs[lane]       + a0 * inv);
    o[lane + 32]  = (_Float16)(hs[lane + 32]  + a1 * inv);
    o[lane + 64]  = (_Float16)(hs[lane + 64]  + a2 * inv);
    o[lane + 96]  = (_Float16)(hs[lane + 96]  + a3 * inv);
}

// ---------------- K10: msg GEMM 32768x128x128 + gelu -> z(f32, d_out) + HOUTH(f16) ----------------
__global__ __launch_bounds__(256) void gemm_msg_kernel(const _Float16* HIN, const _Float16* WmT,
                                                       const float* bmsg, float* zout,
                                                       _Float16* HOUTH) {
    int wave = (blockIdx.x * blockDim.x + threadIdx.x) >> 5;
    int lane = threadIdx.x & 31;
    int r0 = wave * 16;
    const int col = lane & 15, half = lane >> 4;
    v16h a[4];
#pragma unroll
    for (int kk = 0; kk < 4; ++kk) a[kk] = load_a16(HIN + (size_t)r0 * 128 + kk * 32, 128, lane);
    for (int nt = 0; nt < 8; ++nt) {
        v8f c = {};
#pragma unroll
        for (int kk = 0; kk < 4; ++kk)
            c = wmma_f16(a[kk], load_b16(WmT + (size_t)(nt * 16) * 128 + kk * 32, 128, lane), c);
        float bias = bmsg[nt * 16 + col];
#pragma unroll
        for (int v = 0; v < 8; ++v) {
            float x = gelu_f(c[v] + bias);
            size_t idx = (size_t)(r0 + half * 8 + v) * 128 + nt * 16 + col;
            zout[idx] = x;
            HOUTH[idx] = (_Float16)x;
        }
    }
}

// ---------------- K11: coords head ----------------
__global__ __launch_bounds__(256) void coords_kernel(const float* z, const float* coords,
                                                     const float* wcoor, float* co) {
    int row = blockIdx.x * blockDim.x + threadIdx.x;
    if (row >= NTOT) return;
    float a0 = 0.f, a1 = 0.f, a2 = 0.f;
    const float* zr = z + (size_t)row * 128;
    for (int i = 0; i < 128; ++i) {
        float h = zr[i];
        a0 += h * wcoor[i * 3 + 0];
        a1 += h * wcoor[i * 3 + 1];
        a2 += h * wcoor[i * 3 + 2];
    }
    co[(size_t)row * 3 + 0] = coords[(size_t)row * 3 + 0] + tanhf(a0);
    co[(size_t)row * 3 + 1] = coords[(size_t)row * 3 + 1] + tanhf(a1);
    co[(size_t)row * 3 + 2] = coords[(size_t)row * 3 + 2] + tanhf(a2);
}

// ---------------- K12: angles head (A1 WMMA -> LDS -> A2 WMMA -> DyT -> tanh) ----------------
__global__ __launch_bounds__(256) void head_kernel(const _Float16* HOUTH, const _Float16* A1t,
                                                   const float* a1b, const _Float16* A2t,
                                                   const float* a2b, const float* alpha,
                                                   const float* dytw, const float* dytb,
                                                   float* ang) {
    __shared__ __align__(32) _Float16 t1s[8][16 * 128];
    int wslot = threadIdx.x >> 5;
    int lane  = threadIdx.x & 31;
    int wave  = blockIdx.x * 8 + wslot;
    int r0 = wave * 16;
    const int col = lane & 15, half = lane >> 4;

    v16h a[4];
#pragma unroll
    for (int kk = 0; kk < 4; ++kk) a[kk] = load_a16(HOUTH + (size_t)r0 * 128 + kk * 32, 128, lane);
    for (int nt = 0; nt < 7; ++nt) {            // 112 cols (100 valid + zero pad)
        v8f c = {};
#pragma unroll
        for (int kk = 0; kk < 4; ++kk)
            c = wmma_f16(a[kk], load_b16(A1t + (size_t)(nt * 16) * 128 + kk * 32, 128, lane), c);
        int gc = nt * 16 + col;
        float bias = (gc < ENCH) ? a1b[gc] : 0.f;
#pragma unroll
        for (int v = 0; v < 8; ++v)
            t1s[wslot][(half * 8 + v) * 128 + gc] = (_Float16)gelu_f(c[v] + bias);
    }
#pragma unroll
    for (int v = 0; v < 8; ++v)                 // zero pad K = 112..127
        t1s[wslot][(half * 8 + v) * 128 + 112 + col] = (_Float16)0.f;
    __syncthreads();

    v16h a2f[4];
#pragma unroll
    for (int kk = 0; kk < 4; ++kk) a2f[kk] = load_a16(&t1s[wslot][kk * 32], 128, lane);
    v8f c = {};
#pragma unroll
    for (int kk = 0; kk < 4; ++kk)
        c = wmma_f16(a2f[kk], load_b16(A2t + kk * 32, 128, lane), c);
    if (col < OUTC) {
        float al = alpha[0];
        float w = dytw[col], bb = dytb[col], bias = a2b[col];
#pragma unroll
        for (int v = 0; v < 8; ++v) {
            float t = gelu_f(c[v] + bias);
            float tt = tanhf(al * t) * w + bb;
            ang[(size_t)(r0 + half * 8 + v) * OUTC + col] = tanhf(tt);
        }
    }
}

// ---------------- launcher ----------------
extern "C" void kernel_launch(void* const* d_in, const int* in_sizes, int n_in,
                              void* d_out, int out_size, void* d_ws, size_t ws_size,
                              hipStream_t stream) {
    (void)in_sizes; (void)n_in; (void)out_size; (void)ws_size;
    const float* x_res   = (const float*)d_in[0];
    const float* coords  = (const float*)d_in[1];
    const int*   eidx    = (const int*)d_in[2];
    const float* bn_g    = (const float*)d_in[4];
    const float* bn_b    = (const float*)d_in[5];
    const float* W1      = (const float*)d_in[6];
    const float* b1      = (const float*)d_in[7];
    const float* W2      = (const float*)d_in[8];
    const float* b2      = (const float*)d_in[9];
    const float* W3      = (const float*)d_in[10];
    const float* b3      = (const float*)d_in[11];
    const float* embed   = (const float*)d_in[12];
    const float* W_msg   = (const float*)d_in[13];
    const float* b_msg   = (const float*)d_in[14];
    const float* w_coor  = (const float*)d_in[15];
    const float* A1      = (const float*)d_in[16];
    const float* a1      = (const float*)d_in[17];
    const float* A2      = (const float*)d_in[18];
    const float* a2      = (const float*)d_in[19];
    const float* dyt_al  = (const float*)d_in[20];
    const float* dyt_w   = (const float*)d_in[21];
    const float* dyt_b   = (const float*)d_in[22];

    float* out    = (float*)d_out;
    float* out_an = out;
    float* out_z  = out + (size_t)NTOT * OUTC;
    float* out_co = out + (size_t)NTOT * (OUTC + DIM);

    char* ws = (char*)d_ws;
    size_t off = 0;
    auto alloc = [&](size_t bytes) -> void* {
        void* p = ws + off;
        off += (bytes + 255) & ~(size_t)255;
        return p;
    };
    float*    gsum = (float*)alloc(256 * 4);
    float*    gsq  = (float*)alloc(256 * 4);
    float*    scl  = (float*)alloc(256 * 4);
    float*    shf  = (float*)alloc(256 * 4);
    _Float16* W1t  = (_Float16*)alloc((size_t)256 * 256 * 2);
    _Float16* W2t  = (_Float16*)alloc((size_t)128 * 256 * 2);
    _Float16* WmT  = (_Float16*)alloc((size_t)128 * 128 * 2);
    _Float16* A1t  = (_Float16*)alloc((size_t)128 * 128 * 2);
    _Float16* A2t  = (_Float16*)alloc((size_t)16 * 128 * 2);
    _Float16* W3t  = (_Float16*)alloc((size_t)32 * 128 * 2);
    int*      ids  = (int*)alloc((size_t)NTOT * 4);
    _Float16* HOUTH= (_Float16*)alloc((size_t)NTOT * DIM * 2);
    char*     big0 = (char*)alloc((size_t)NTOT * INC * 2);   // XN, later HF (f32 DIM)
    char*     big1 = (char*)alloc((size_t)NTOT * INC * 2);   // H1, later MASK+HIN
    _Float16* XN   = (_Float16*)big0;
    float*    HF   = (float*)big0;                 // NTOT*128 f32 == same bytes
    _Float16* H1   = (_Float16*)big1;
    unsigned* MASK = (unsigned*)big1;              // NTOT*64 u32 = 8.4 MB
    _Float16* HIN  = (_Float16*)(big1 + (size_t)NTOT * 64 * 4);

    prep_kernel<<<538, 256, 0, stream>>>(W1, W2, W_msg, A1, A2, W3,
                                         W1t, W2t, WmT, A1t, A2t, W3t, gsum, gsq);
    colsum_kernel<<<256, 256, 0, stream>>>(x_res, gsum, gsq);
    finalize_kernel<<<1, 256, 0, stream>>>(gsum, gsq, bn_g, bn_b, scl, shf);
    bncast_kernel<<<(NTOT * INC) / 256, 256, 0, stream>>>(x_res, scl, shf, XN);
    gemm1_kernel<<<256, 256, 0, stream>>>(XN, W1t, b1, H1);               // 2048 waves
    gemm23_argmax_kernel<<<256, 256, 0, stream>>>(H1, W2t, b2, W3t, b3, ids);
    embed_kernel<<<(NTOT * DIM) / 256, 256, 0, stream>>>(ids, embed, HF);
    zero_mask_kernel<<<(NTOT * 64 / 4) / 256, 256, 0, stream>>>((uint4*)MASK, NTOT * 64 / 4);
    scatter_kernel<<<NEDGE / 256, 256, 0, stream>>>(eidx, MASK);
    aggregate_kernel<<<(NTOT * 32) / 256, 256, 0, stream>>>(MASK, HF, HIN);
    gemm_msg_kernel<<<256, 256, 0, stream>>>(HIN, WmT, b_msg, out_z, HOUTH);
    coords_kernel<<<NTOT / 256, 256, 0, stream>>>(out_z, coords, w_coor, out_co);
    head_kernel<<<256, 256, 0, stream>>>(HOUTH, A1t, a1, A2t, a2, dyt_al, dyt_w, dyt_b, out_an);
}
